// ContextLSTM_360777253535
// MI455X (gfx1250) — matrix operand order repaired and checked by reading, not compiled, not run on hardware
//
#include <hip/hip_runtime.h>

// ===================== Problem constants =====================
#define L_FEAT   128
#define INP      384              // T_IN * L_FEAT
#define HID      640              // T_OUT * L_FEAT
#define NGATE    4
#define BATCH    32
#define MAX_T    1024
#define T_OUTN   1022             // MAX_T - (T_OUT - T_IN)
#define KTOT     1024             // INP + HID
#define KITERS   (KTOT / 32)      // 32 bf16 WMMA k-steps
#define NSTRIPS  (HID / 16)       // 40 hidden-unit strips
#define WPB      2                // waves per block (LDS-limited: 2*128KB = 256KB)
#define NBLOCKS  (NSTRIPS / WPB)  // 20 persistent blocks
#define THREADS  (WPB * 32)       // wave32
#define LDS_PER_WAVE (KITERS * NGATE * 32 * 32)  // 131072 B of bf16 B-fragments
#define LDS_BYTES    (WPB * LDS_PER_WAVE)        // 262144 B (<320KB WGP LDS)

typedef __bf16 v16bf __attribute__((ext_vector_type(16)));
typedef float  v8f   __attribute__((ext_vector_type(8)));

union Frag { v16bf v; uint4 u[2]; };

__device__ __forceinline__ unsigned short f32_bf16(float f) {
    unsigned u = __float_as_uint(f);
    unsigned r = 0x7FFFu + ((u >> 16) & 1u);   // round-to-nearest-even
    return (unsigned short)((u + r) >> 16);
}
__device__ __forceinline__ float sigm(float x) { return 1.0f / (1.0f + __expf(-x)); }
__device__ __forceinline__ float tanh_(float x) { return tanhf(x); }

// Grid-wide barrier: monotonic counter, agent scope (persistent kernel, 20 blocks).
__device__ __forceinline__ void grid_barrier(unsigned* bar, unsigned gen) {
    __syncthreads();
    if (threadIdx.x == 0) {
        __threadfence();
        __hip_atomic_fetch_add(bar, 1u, __ATOMIC_ACQ_REL, __HIP_MEMORY_SCOPE_AGENT);
        while (__hip_atomic_load(bar, __ATOMIC_ACQUIRE, __HIP_MEMORY_SCOPE_AGENT) <
               gen * (unsigned)NBLOCKS) {
            __builtin_amdgcn_s_sleep(2);
        }
    }
    __syncthreads();
}

// Per-call init: zero barrier + both ping-pong [x|h] buffers, emit n_batches tail
// of the tuple output. Keeps kernel_launch deterministic across graph replays.
__global__ void ctxlstm_init(const int* __restrict__ nb, int* __restrict__ out_tail,
                             unsigned short* __restrict__ xh, unsigned* __restrict__ bar) {
    int tid = blockIdx.x * blockDim.x + threadIdx.x;
    int stride = gridDim.x * blockDim.x;
    if (tid == 0) *bar = 0u;
    for (int i = tid; i < 2 * BATCH * KTOT; i += stride) xh[i] = 0;
    for (int i = tid; i < T_OUTN; i += stride) out_tail[i] = nb[i];
}

__global__ void __launch_bounds__(THREADS, 1)
ctxlstm_persistent(const float* __restrict__ pad_seq,
                   const float* __restrict__ W_ih, const float* __restrict__ W_hh,
                   const float* __restrict__ b_ih, const float* __restrict__ b_hh,
                   const int* __restrict__ n_batches,
                   float* __restrict__ y,
                   unsigned short* __restrict__ xh0,   // ping-pong (2, 32, 1024) bf16
                   unsigned* __restrict__ bar) {
    extern __shared__ unsigned char lds_raw[];
    const int lane   = threadIdx.x & 31;
    const int wib    = threadIdx.x >> 5;
    const int strip  = blockIdx.x * WPB + wib;       // 0..39 : 16 hidden units each
    const int col16  = lane & 15;
    const int khalf  = (lane >> 4) & 1;              // which K-half this lane carries
    const int ncol   = strip * 16 + col16;           // hidden-unit index of this lane
    unsigned short* wlds = (unsigned short*)(lds_raw + wib * LDS_PER_WAVE);

    // ---- Prepack fp32 weights -> bf16 WMMA B-fragments in LDS (once) ----
    // B is K x N with B[k,n] = W[n,k]. Lane L holds N = L%16, K-half = (L>=16).
    for (int ki = 0; ki < KITERS; ++ki) {
        for (int g = 0; g < NGATE; ++g) {
            const int n = g * HID + ncol;
            unsigned short* slot = wlds + (((ki * NGATE + g) * 32) + lane) * 16;
            const int kbase = ki * 32 + khalf * 16;
            #pragma unroll
            for (int j = 0; j < 16; ++j) {
                const int K = kbase + j;
                const float w = (K < INP) ? W_ih[n * INP + K]
                                          : W_hh[n * HID + (K - INP)];
                slot[j] = f32_bf16(w);
            }
        }
    }

    // Bias folded into WMMA accumulator init (per-column, same for all C rows).
    float bias[NGATE];
    #pragma unroll
    for (int g = 0; g < NGATE; ++g)
        bias[g] = b_ih[g * HID + ncol] + b_hh[g * HID + ncol];

    // Cell/hidden state in WMMA C-layout registers: rows owned by this lane.
    float cst[2][8], hst[2][8];
    #pragma unroll
    for (int mt = 0; mt < 2; ++mt)
        #pragma unroll
        for (int r = 0; r < 8; ++r) { cst[mt][r] = 0.0f; hst[mt][r] = 0.0f; }

    const int gtid    = blockIdx.x * THREADS + threadIdx.x;
    const int gstride = NBLOCKS * THREADS;

    // Prologue: stage x window for t=0 into buffer 0 (h region already zeroed).
    // Vectorized: float4 -> 4 x bf16 -> one 8B store.
    for (int i = gtid; i < BATCH * (INP / 4); i += gstride) {
        const int m = i / (INP / 4), j4 = (i % (INP / 4)) * 4;
        const float4 f = *(const float4*)(pad_seq + (m * MAX_T + 0) * L_FEAT + j4);
        ushort4 o;
        o.x = f32_bf16(f.x); o.y = f32_bf16(f.y);
        o.z = f32_bf16(f.z); o.w = f32_bf16(f.w);
        *(ushort4*)(xh0 + m * KTOT + j4) = o;
    }
    unsigned gen = 0;
    grid_barrier(bar, ++gen);

    for (int t = 0; t < T_OUTN; ++t) {
        // Direct offset arithmetic from the kernel-arg pointer keeps these in
        // the GLOBAL address space (global_load_b128, decoupled from DScnt).
        const unsigned short* cur = xh0 + (t & 1) * (BATCH * KTOT);
        unsigned short*       nxt = xh0 + ((t + 1) & 1) * (BATCH * KTOT);

        // ---- GEMM: gates[32 x 64(this wave)] = [x|h] @ W^T + bias, bf16 WMMA ----
        v8f acc[NGATE][2];
        #pragma unroll
        for (int g = 0; g < NGATE; ++g) {
            const v8f bsplat = {bias[g], bias[g], bias[g], bias[g],
                                bias[g], bias[g], bias[g], bias[g]};
            acc[g][0] = bsplat;
            acc[g][1] = bsplat;
        }

        #pragma unroll 4
        for (int ki = 0; ki < KITERS; ++ki) {
            const int k0 = ki * 32 + khalf * 16;     // A-layout K chunks: k0, k0+8 (contiguous 16)
            Frag a0, a1;
            a0.u[0] = *(const uint4*)(cur + (col16 +  0) * KTOT + k0);
            a0.u[1] = *(const uint4*)(cur + (col16 +  0) * KTOT + k0 + 8);
            a1.u[0] = *(const uint4*)(cur + (col16 + 16) * KTOT + k0);
            a1.u[1] = *(const uint4*)(cur + (col16 + 16) * KTOT + k0 + 8);
            #pragma unroll
            for (int g = 0; g < NGATE; ++g) {
                Frag b;
                const uint4* bp =
                    (const uint4*)(wlds + (((ki * NGATE + g) * 32) + lane) * 16);
                b.u[0] = bp[0];
                b.u[1] = bp[1];
                acc[g][0] = __builtin_amdgcn_wmma_f32_16x16x32_bf16(
                    false, a0.v, false, b.v, (short)0, acc[g][0], false, false);
                acc[g][1] = __builtin_amdgcn_wmma_f32_16x16x32_bf16(
                    false, a1.v, false, b.v, (short)0, acc[g][1], false, false);
            }
        }

        // ---- Elementwise LSTM cell for this wave's 16 hidden units ----
        const int nbv = n_batches[t];
        #pragma unroll
        for (int mt = 0; mt < 2; ++mt) {
            #pragma unroll
            for (int r = 0; r < 8; ++r) {
                const int row = mt * 16 + khalf * 8 + r;   // C-layout row of this lane
                const float ig = acc[0][mt][r];
                const float fg = acc[1][mt][r];
                const float gg = acc[2][mt][r];
                const float og = acc[3][mt][r];
                const float cn = sigm(fg) * cst[mt][r] + sigm(ig) * tanh_(gg);
                const float hn = sigm(og) * tanh_(cn);
                const bool act = row < nbv;
                const float h2 = act ? hn : hst[mt][r];
                const float c2 = act ? cn : cst[mt][r];
                hst[mt][r] = h2;
                cst[mt][r] = c2;
                // carry h into next ping-pong buffer (all rows, so state persists)
                nxt[row * KTOT + INP + ncol] = f32_bf16(h2);
                y[(row * T_OUTN + t) * HID + ncol] = act ? hn : 0.0f;
            }
        }

        // ---- Stage next x window into the other buffer (cooperative) ----
        if (t + 1 < T_OUTN) {
            for (int i = gtid; i < BATCH * (INP / 4); i += gstride) {
                const int m = i / (INP / 4), j4 = (i % (INP / 4)) * 4;
                const float4 f =
                    *(const float4*)(pad_seq + (m * MAX_T + (t + 1)) * L_FEAT + j4);
                ushort4 o;
                o.x = f32_bf16(f.x); o.y = f32_bf16(f.y);
                o.z = f32_bf16(f.z); o.w = f32_bf16(f.w);
                *(ushort4*)(nxt + m * KTOT + j4) = o;
                if (t + 2 < T_OUTN)
                    __builtin_prefetch(pad_seq + (m * MAX_T + (t + 2)) * L_FEAT + j4, 0, 1);
            }
        }
        __threadfence();
        grid_barrier(bar, ++gen);   // single barrier per step (ping-pong removes WAR)
    }
}

extern "C" void kernel_launch(void* const* d_in, const int* in_sizes, int n_in,
                              void* d_out, int out_size, void* d_ws, size_t ws_size,
                              hipStream_t stream) {
    const float* pad_seq   = (const float*)d_in[0];
    const float* W_ih      = (const float*)d_in[1];
    const float* W_hh      = (const float*)d_in[2];
    const float* b_ih      = (const float*)d_in[3];
    const float* b_hh      = (const float*)d_in[4];
    const int*   n_batches = (const int*)d_in[5];

    float* y        = (float*)d_out;
    int*   out_tail = (int*)d_out + BATCH * T_OUTN * HID;   // tuple: (y, n_batches)

    unsigned*       bar = (unsigned*)d_ws;
    unsigned short* xh  = (unsigned short*)((char*)d_ws + 256);  // 2 x 32 x 1024 bf16

    hipLaunchKernelGGL(ctxlstm_init, dim3(8), dim3(256), 0, stream,
                       n_batches, out_tail, xh, bar);
    hipLaunchKernelGGL(ctxlstm_persistent, dim3(NBLOCKS), dim3(THREADS), LDS_BYTES,
                       stream, pad_seq, W_ih, W_hh, b_ih, b_hh, n_batches, y, xh, bar);
}
